// DGC_9818295239337
// MI455X (gfx1250) — matrix operand order repaired and checked by reading, not compiled, not run on hardware
//
#include <hip/hip_runtime.h>

#define NN 16384
#define DD 128
#define KB 64          // K elements consumed per outer iteration (2 WMMAs)
#define ROWS 16        // M-tile rows per workgroup
#define THREADS 256    // 8 wave32
#define APITCH 72      // bf16 elements per LDS A row (64 + 8 pad -> conflict-free b128)

typedef __attribute__((ext_vector_type(16))) __bf16 v16bf;
typedef __attribute__((ext_vector_type(8)))  __bf16 v8bf;
typedef __attribute__((ext_vector_type(8)))  float  v8f;

static __device__ __forceinline__ __bf16 f2bf(float f) {
  unsigned u = __builtin_bit_cast(unsigned, f);
  u += 0x7FFFu + ((u >> 16) & 1u);            // round-to-nearest-even
  unsigned short s = (unsigned short)(u >> 16);
  return __builtin_bit_cast(__bf16, s);
}

// Prepass: hT[d][n] = bf16(h[n][d]); coalesced on the write side.
__global__ __launch_bounds__(256)
void DGC_hT_kernel(const float* __restrict__ h, __bf16* __restrict__ hT) {
  int idx = blockIdx.x * 256 + threadIdx.x;   // 0 .. DD*NN-1
  int d = idx >> 14;                          // / NN
  int n = idx & (NN - 1);
  hT[(size_t)d * NN + n] = f2bf(h[(size_t)n * DD + d]);
}

// Fused row-softmax(adj) @ h using bf16 WMMA with f32 accumulation.
__global__ __launch_bounds__(THREADS)
void DGC_softmax_mm_kernel(const float* __restrict__ adj,
                           const __bf16* __restrict__ hT,
                           float* __restrict__ out) {
  __shared__ __bf16 Abuf[2][ROWS][APITCH];
  __shared__ float  partial[THREADS];
  __shared__ float  rowsum[ROWS];

  const int tid  = threadIdx.x;
  const int row0 = blockIdx.x * ROWS;

  // cooperative-load role: thread -> (row r, 4-float column group cg)
  const int r  = tid >> 4;                    // 0..15
  const int cg = tid & 15;                    // 0..15 -> cols 4*cg..4*cg+3

  // WMMA role: wave owns one 16-wide N-tile of the D=128 output
  const int wave = tid >> 5;                  // 0..7
  const int lane = tid & 31;
  const int nn   = lane & 15;                 // N (and A-matrix M) index
  const int hi   = lane >> 4;                 // lane-half selector

  const __bf16* __restrict__ bcol = hT + (size_t)(wave * 16 + nn) * NN;

  v8f c = {};                                 // f32 C/D accumulator
  float rsum = 0.f;

  for (int kk = 0; kk < NN; kk += KB) {
    const int buf = (kk >> 6) & 1;

    // ---- load 16x64 adj tile, exp, row-sum, stash bf16 in LDS ----
    float4 av = *(const float4*)(adj + (size_t)(row0 + r) * NN + kk + cg * 4);
    float e0 = __expf(av.x), e1 = __expf(av.y);
    float e2 = __expf(av.z), e3 = __expf(av.w);
    rsum += (e0 + e1) + (e2 + e3);
    __bf16* dst = &Abuf[buf][r][cg * 4];
    dst[0] = f2bf(e0); dst[1] = f2bf(e1);
    dst[2] = f2bf(e2); dst[3] = f2bf(e3);
    __syncthreads();

    // ---- two 16x16x32 bf16 WMMAs over this K-chunk ----
#pragma unroll
    for (int kb = 0; kb < KB; kb += 32) {
      // A fragment (contiguous K order, matching B):
      // low lanes: K 0-15; high lanes: K 16-31
      v8bf alo = *(const v8bf*)&Abuf[buf][nn][kb + hi * 16];
      v8bf ahi = *(const v8bf*)&Abuf[buf][nn][kb + hi * 16 + 8];
      v16bf a = __builtin_shufflevector(alo, ahi,
                                        0, 1, 2, 3, 4, 5, 6, 7,
                                        8, 9, 10, 11, 12, 13, 14, 15);
      // B fragment: lane n holds K (kb + hi*16) .. +15 of column n — contiguous in hT
      v16bf b = *(const v16bf*)(bcol + kk + kb + hi * 16);
      c = __builtin_amdgcn_wmma_f32_16x16x32_bf16(
              /*neg_a=*/false, a, /*neg_b=*/false, b,
              /*c_mod=*/(short)0, c, /*reuse_a=*/false, /*reuse_b=*/false);
    }
  }

  // ---- reduce row sums across the 16 threads per row ----
  partial[tid] = rsum;
  __syncthreads();
  if (tid < ROWS) {
    float s = 0.f;
#pragma unroll
    for (int j = 0; j < 16; ++j) s += partial[tid * 16 + j];
    rowsum[tid] = s;
  }
  __syncthreads();

  // ---- normalize + store: VGPR j holds (M = j + 8*hi, N = nn) ----
  float* __restrict__ orow = out + (size_t)row0 * DD + wave * 16 + nn;
#pragma unroll
  for (int j = 0; j < 8; ++j) {
    int m = hi * 8 + j;
    orow[(size_t)m * DD] = c[j] / rowsum[m];
  }
}

extern "C" void kernel_launch(void* const* d_in, const int* in_sizes, int n_in,
                              void* d_out, int out_size, void* d_ws, size_t ws_size,
                              hipStream_t stream) {
  const float* h   = (const float*)d_in[0];   // [NN, DD] f32
  const float* adj = (const float*)d_in[1];   // [NN, NN] f32
  float* out = (float*)d_out;                 // [NN, DD] f32
  __bf16* hT = (__bf16*)d_ws;                 // [DD, NN] bf16 = 4 MiB scratch

  DGC_hT_kernel<<<(NN * DD) / 256, 256, 0, stream>>>(h, hT);
  DGC_softmax_mm_kernel<<<NN / ROWS, THREADS, 0, stream>>>(adj, hT, out);
}
